// LSTMModel_51221779972742
// MI455X (gfx1250) — matrix operand order repaired and checked by reading, not compiled, not run on hardware
//
#include <hip/hip_runtime.h>
#include <cstdint>
#include <cstddef>

// ---------------------------------------------------------------------------
// LSTM stack for MI455X (gfx1250), wave32 + WMMA bf16.
//
// y = head(h2[:, T-1]) => per-timestep work is only the two recurrences.
// Pipeline:
//   1) prep: weights->bf16, bias combine, x -> bf16 [T,B,32]
//   2) lstm1: in-loop input proj; ALL 12 weight B-frags pre-loaded into
//      registers before the t-loop (no per-step weight traffic)
//   3) xg2 GEMM (parallel over B*T): xg2 = h1 @ Wih2^T + bias2, stored
//      pre-swizzled in WMMA C/D fragment layout (bf16, 16B/lane/gate)
//   4) lstm2: only h @ Whh^T in the serial loop — 16 WMMAs/step with all
//      16 B-frags (128 VGPRs) register-resident; xg fragments streamed with
//      register double-buffering; cell update in registers (v_tanh_f32);
//      h double-buffered in LDS; ONE barrier per step.
//   5) head on the final h2 only.
//
// Wave->tile mapping: wave wv owns one 16-wide j-slice replicated across the
// four gate blocks, so i,f,g,o for any (m,j) are in the same lane/slot of the
// 4 accumulators (cell update fully in registers).
// ---------------------------------------------------------------------------

typedef __attribute__((ext_vector_type(16))) __bf16 v16bf;
typedef __attribute__((ext_vector_type(8)))  float  v8f;
using u16 = unsigned short;

union Frag { v16bf v; uint4 q[2]; };
union Acc  { v8f v; float f[8]; };

constexpr int Bc = 512, Tc = 1024, DINc = 32, H1c = 64, H2c = 128, H3c = 256;
constexpr int NTHREADS = 256;   // 8 waves of 32
constexpr int NW = 8;
constexpr int MT = 16;          // batch rows per workgroup (WMMA M)
constexpr int NB = Bc / MT;     // 32 batch tiles

static __device__ __forceinline__ u16 f32_to_bf16(float f) {
    union { float f; uint32_t u; } v; v.f = f;
    uint32_t u = v.u;
    uint32_t r = 0x7FFFu + ((u >> 16) & 1u);   // round-to-nearest-even
    return (u16)((u + r) >> 16);
}

static __device__ __forceinline__ float bf16_to_f32(u16 b) {
    union { uint32_t u; float f; } v; v.u = ((uint32_t)b) << 16;
    return v.f;
}

static __device__ __forceinline__ float fast_tanh(float x) {
#if __has_builtin(__builtin_amdgcn_tanhf)
    return __builtin_amdgcn_tanhf(x);          // V_TANH_F32 (confirmed lowers)
#elif __has_builtin(__builtin_amdgcn_tanh_f32)
    return __builtin_amdgcn_tanh_f32(x);
#else
    const float e = __expf(2.0f * x);
    return 1.0f - 2.0f * __builtin_amdgcn_rcpf(e + 1.0f);
#endif
}

static __device__ __forceinline__ float fast_sigmoid(float x) {
    return 0.5f * fast_tanh(0.5f * x) + 0.5f;  // single trans op
}

static __device__ __forceinline__ Frag load_bfrag(const u16* p) {
    Frag b;
    const int hh = (threadIdx.x >> 4) & 1;
    b.q[0] = *(const uint4*)(p - hh * 8);
    b.q[1] = *(const uint4*)(p + 16 - hh * 8);
    return b;
}

// ---------------------------------------------------------------------------
// Layer 1: HIN=32, H=64. Input projection in-loop; all 12 weight B-frags
// pre-loaded to registers. x A-fragments come straight from global
// (16B-contiguous chunks in [T,B,32]), register double-buffered.
// LDS holds only double-buffered h.
// ---------------------------------------------------------------------------
__global__ __launch_bounds__(NTHREADS, 1)
void lstm1_kernel(const u16*   __restrict__ xin,   // [T, B, 32] bf16
                  const u16*   __restrict__ Wih,   // [256, 32]  bf16
                  const u16*   __restrict__ Whh,   // [256, 64]  bf16
                  const float* __restrict__ bias,  // [256] (bih+bhh)
                  u16*         __restrict__ hout)  // [T, B, 64] bf16
{
    constexpr int HIN = DINc, H = H1c;
    constexpr int AW = H / 16;          // 4 active WMMA waves
    constexpr int KH = H / 32;          // 2 recurrent K-steps (KX = 1)

    __shared__ u16 sh_h[2][MT * H];

    const int tid = threadIdx.x, lane = tid & 31, wv = tid >> 5;
    const int bo = blockIdx.x * MT;
    const int mA = lane & 15, hA = lane >> 4;
    const int nB = lane & 15, kh = lane >> 4;
    const int jc = wv * 16 + nB;

    for (int e = tid; e < 2 * MT * H; e += NTHREADS) ((u16*)sh_h)[e] = 0;

    // ---- register-resident weights & bias (explicit pre-load) ----
    float bia[4] = {0.f, 0.f, 0.f, 0.f};
    Frag bwih[4];           // input-proj B-frags (K=32)
    Frag bwhh[KH][4];       // recurrent B-frags
    if (wv < AW) {
        #pragma unroll
        for (int g = 0; g < 4; ++g) {
            bia[g]  = bias[g * H + jc];
            bwih[g] = load_bfrag(Wih + (size_t)(g * H + jc) * HIN + kh * 16);
            #pragma unroll
            for (int k = 0; k < KH; ++k)
                bwhh[k][g] = load_bfrag(Whh + (size_t)(g * H + jc) * H + k * 32 + kh * 16);
        }
    }

    Acc cst;                            // c-state in registers
    #pragma unroll
    for (int r = 0; r < 8; ++r) cst.f[r] = 0.0f;

    // A-fragment of x for t=0 (KX == 1): two contiguous 16B chunks per lane
    Frag xfr;
    if (wv < AW) {
        const u16* pa = xin + ((size_t)0 * Bc + bo + mA) * HIN + hA * 8;
        xfr.q[0] = *(const uint4*)(pa);
        xfr.q[1] = *(const uint4*)(pa + 16);
    }

    for (int t = 0; t < Tc; ++t) {
        __syncthreads();   // sh_h[t&1] complete

        // prefetch next x A-fragment (consumed next iteration)
        Frag xnx;
        const bool pf = (t + 1 < Tc) && (wv < AW);
        if (pf) {
            const u16* pa = xin + ((size_t)(t + 1) * Bc + bo + mA) * HIN + hA * 8;
            xnx.q[0] = *(const uint4*)(pa);
            xnx.q[1] = *(const uint4*)(pa + 16);
        }

        if (wv < AW) {   // wave-uniform; EXEC all-ones inside
            const u16* hb = sh_h[t & 1];
            u16*       hn = sh_h[(t + 1) & 1];

            Acc acc[4];
            #pragma unroll
            for (int g = 0; g < 4; ++g) {
                #pragma unroll
                for (int r = 0; r < 8; ++r) acc[g].f[r] = bia[g];
            }

            // input projection (K = 32) — weights already in VGPRs
            #pragma unroll
            for (int g = 0; g < 4; ++g)
                acc[g].v = __builtin_amdgcn_wmma_f32_16x16x32_bf16(
                    false, xfr.v, false, bwih[g].v, (short)0, acc[g].v, false, false);

            // recurrent part — weights already in VGPRs
            #pragma unroll
            for (int k = 0; k < KH; ++k) {
                Frag a;
                const u16* pa = hb + mA * H + k * 32 + hA * 8;
                a.q[0] = *(const uint4*)(pa);
                a.q[1] = *(const uint4*)(pa + 16);
                #pragma unroll
                for (int g = 0; g < 4; ++g)
                    acc[g].v = __builtin_amdgcn_wmma_f32_16x16x32_bf16(
                        false, a.v, false, bwhh[k][g].v, (short)0, acc[g].v, false, false);
            }

            // in-register LSTM cell update
            #pragma unroll
            for (int r = 0; r < 8; ++r) {
                const int m  = r + 8 * hA;
                const float gi = fast_sigmoid(acc[0].f[r]);
                const float gf = fast_sigmoid(acc[1].f[r]);
                const float gg = fast_tanh   (acc[2].f[r]);
                const float go = fast_sigmoid(acc[3].f[r]);
                const float c  = gf * cst.f[r] + gi * gg;
                const float h  = go * fast_tanh(c);
                cst.f[r] = c;
                const u16 hbv = f32_to_bf16(h);
                hn[m * H + jc] = hbv;
                hout[((size_t)t * Bc + bo + m) * H + jc] = hbv;
            }
        }
        xfr = xnx;   // rotate prefetch register
    }
}

// ---------------------------------------------------------------------------
// xg2 GEMM (parallel over all B*T rows): xg2 = h1 @ Wih2^T + bias2.
// Output stored pre-swizzled per WMMA fragment:
//   xg[((mt*NW + wv)*4 + g)*256 + lane*8 .. +7]  (bf16)
// A-fragments load straight from global; no LDS, no barriers.
// ---------------------------------------------------------------------------
__global__ __launch_bounds__(NTHREADS)
void xg2_gemm_kernel(const u16*   __restrict__ h1,    // [T*B, 64] bf16
                     const u16*   __restrict__ Wih,   // [512, 64] bf16
                     const float* __restrict__ bias,  // [512] (bih+bhh)
                     u16*         __restrict__ xg)    // fragment layout
{
    constexpr int HIN = H1c, H = H2c;   // 64 -> 512 gates
    constexpr int KX = HIN / 32;        // 2

    const int tid = threadIdx.x, lane = tid & 31, wv = tid >> 5;
    const size_t mt = blockIdx.x;       // M-tile over T*B/16
    const size_t rowbase = mt * MT;
    const int mA = lane & 15, hA = lane >> 4;
    const int nB = lane & 15, kh = lane >> 4;
    const int jc = wv * 16 + nB;

    Acc acc[4];
    #pragma unroll
    for (int g = 0; g < 4; ++g) {
        const float bv = bias[g * H + jc];
        #pragma unroll
        for (int r = 0; r < 8; ++r) acc[g].f[r] = bv;
    }

    #pragma unroll
    for (int k = 0; k < KX; ++k) {
        Frag a;
        const u16* pa = h1 + (rowbase + mA) * HIN + k * 32 + hA * 8;
        a.q[0] = *(const uint4*)(pa);
        a.q[1] = *(const uint4*)(pa + 16);
        #pragma unroll
        for (int g = 0; g < 4; ++g) {
            Frag b = load_bfrag(Wih + (size_t)(g * H + jc) * HIN + k * 32 + kh * 16);
            acc[g].v = __builtin_amdgcn_wmma_f32_16x16x32_bf16(
                false, a.v, false, b.v, (short)0, acc[g].v, false, false);
        }
    }

    #pragma unroll
    for (int g = 0; g < 4; ++g) {
        u16 tmp[8];
        #pragma unroll
        for (int r = 0; r < 8; ++r) tmp[r] = f32_to_bf16(acc[g].f[r]);
        *(uint4*)(xg + (((mt * NW + wv) * 4 + g) * 256) + lane * 8) =
            *(const uint4*)tmp;
    }
}

// ---------------------------------------------------------------------------
// Layer 2 recurrence: only h @ Whh^T in the serial loop. 16 WMMAs/step;
// all 16 B-frags (128 VGPRs) pre-loaded; xg register-double-buffered.
// ---------------------------------------------------------------------------
__global__ __launch_bounds__(NTHREADS, 1)
void lstm2_kernel(const u16*   __restrict__ xg,    // swizzled fragments (bf16)
                  const u16*   __restrict__ Whh,   // [512, 128] bf16
                  float*       __restrict__ hlast) // [B, 128] f32
{
    constexpr int H = H2c;              // 128: all 8 waves active
    constexpr int KH = H / 32;          // 4 recurrent K-steps

    __shared__ u16 sh_h[2][MT * H];

    const int tid = threadIdx.x, lane = tid & 31, wv = tid >> 5;
    const int bt = blockIdx.x;
    const int bo = bt * MT;
    const int mA = lane & 15, hA = lane >> 4;
    const int nB = lane & 15, kh = lane >> 4;
    const int jc = wv * 16 + nB;

    for (int e = tid; e < 2 * MT * H; e += NTHREADS) ((u16*)sh_h)[e] = 0;

    // ---- register-resident recurrent weights (explicit pre-load) ----
    Frag bw[KH][4];   // 16 frags = 128 VGPRs
    #pragma unroll
    for (int k = 0; k < KH; ++k)
        #pragma unroll
        for (int g = 0; g < 4; ++g)
            bw[k][g] = load_bfrag(Whh + (size_t)(g * H + jc) * H + k * 32 + kh * 16);

    Acc cst;
    #pragma unroll
    for (int r = 0; r < 8; ++r) cst.f[r] = 0.0f;

    auto xg_ptr = [&](int t, int g) -> const uint4* {
        return (const uint4*)(xg +
            ((((size_t)t * NB + bt) * NW + wv) * 4 + g) * 256 + lane * 8);
    };

    uint4 xcur[4];
    #pragma unroll
    for (int g = 0; g < 4; ++g) xcur[g] = *xg_ptr(0, g);

    for (int t = 0; t < Tc; ++t) {
        __syncthreads();   // sh_h[t&1] complete

        // prefetch next xg fragment (consumed next iteration)
        uint4 xnx[4];
        if (t + 1 < Tc) {
            #pragma unroll
            for (int g = 0; g < 4; ++g) xnx[g] = *xg_ptr(t + 1, g);
        }

        const u16* hb = sh_h[t & 1];
        u16*       hn = sh_h[(t + 1) & 1];

        Acc acc[4];
        #pragma unroll
        for (int g = 0; g < 4; ++g) {
            #pragma unroll
            for (int r = 0; r < 8; ++r) acc[g].f[r] = 0.0f;
        }

        #pragma unroll
        for (int k = 0; k < KH; ++k) {
            Frag a;
            const u16* pa = hb + mA * H + k * 32 + hA * 8;
            a.q[0] = *(const uint4*)(pa);
            a.q[1] = *(const uint4*)(pa + 16);
            #pragma unroll
            for (int g = 0; g < 4; ++g)
                acc[g].v = __builtin_amdgcn_wmma_f32_16x16x32_bf16(
                    false, a.v, false, bw[k][g].v, (short)0, acc[g].v, false, false);
        }

        // add precomputed input projection (+bias): bf16 -> f32 is <<16
        #pragma unroll
        for (int g = 0; g < 4; ++g) {
            const u16* p = (const u16*)&xcur[g];
            #pragma unroll
            for (int r = 0; r < 8; ++r) acc[g].f[r] += bf16_to_f32(p[r]);
        }

        // in-register LSTM cell update
        #pragma unroll
        for (int r = 0; r < 8; ++r) {
            const int m  = r + 8 * hA;
            const float gi = fast_sigmoid(acc[0].f[r]);
            const float gf = fast_sigmoid(acc[1].f[r]);
            const float gg = fast_tanh   (acc[2].f[r]);
            const float go = fast_sigmoid(acc[3].f[r]);
            const float c  = gf * cst.f[r] + gi * gg;
            const float h  = go * fast_tanh(c);
            cst.f[r] = c;
            hn[m * H + jc] = f32_to_bf16(h);
            if (t == Tc - 1) hlast[(size_t)(bo + m) * H + jc] = h;
        }

        #pragma unroll
        for (int g = 0; g < 4; ++g) xcur[g] = xnx[g];
    }
}

// ---------------------------------------------------------------------------
// FC head on the final timestep only: y = W2 @ relu(W1 @ h + b1) + b2
// ---------------------------------------------------------------------------
__global__ __launch_bounds__(H3c)
void head_kernel(const float* __restrict__ h2last,  // [B, H2]
                 const float* __restrict__ W1,      // [H3, H2]
                 const float* __restrict__ b1,      // [H3]
                 const float* __restrict__ W2,      // [1, H3]
                 const float* __restrict__ b2,      // [1]
                 float* __restrict__ y)             // [B]
{
    __shared__ float red[H3c];
    const int b = blockIdx.x, j = threadIdx.x;
    const float* h = h2last + (size_t)b * H2c;
    const float* w = W1 + (size_t)j * H2c;
    float s = 0.0f;
    #pragma unroll 4
    for (int k = 0; k < H2c; ++k) s = fmaf(w[k], h[k], s);
    const float z = fmaxf(s + b1[j], 0.0f);
    red[j] = z * W2[j];
    __syncthreads();
    for (int off = H3c / 2; off > 0; off >>= 1) {
        if (j < off) red[j] += red[j + off];
        __syncthreads();
    }
    if (j == 0) y[b] = red[0] + b2[0];
}

// ---------------------------------------------------------------------------
// Prep kernels
// ---------------------------------------------------------------------------
__global__ void cvt_bf16_kernel(const float* __restrict__ in,
                                u16* __restrict__ out, int n) {
    for (int i = blockIdx.x * blockDim.x + threadIdx.x; i < n;
         i += gridDim.x * blockDim.x)
        out[i] = f32_to_bf16(in[i]);
}

__global__ void add_bias_kernel(const float* __restrict__ a,
                                const float* __restrict__ b,
                                float* __restrict__ out, int n) {
    for (int i = blockIdx.x * blockDim.x + threadIdx.x; i < n;
         i += gridDim.x * blockDim.x)
        out[i] = a[i] + b[i];
}

// x [B,T,DIN] f32 -> bf16 [T,B,DIN]; output-index coalesced
__global__ void xpose_cvt_kernel(const float* __restrict__ in,
                                 u16* __restrict__ out) {
    const size_t n = (size_t)Bc * Tc * DINc;
    for (size_t o = (size_t)blockIdx.x * blockDim.x + threadIdx.x; o < n;
         o += (size_t)gridDim.x * blockDim.x) {
        const int    j = (int)(o % DINc);
        const size_t r = o / DINc;
        const int    b = (int)(r % Bc);
        const int    t = (int)(r / Bc);
        out[o] = f32_to_bf16(in[((size_t)b * Tc + t) * DINc + j]);
    }
}

// ---------------------------------------------------------------------------
extern "C" void kernel_launch(void* const* d_in, const int* in_sizes, int n_in,
                              void* d_out, int out_size, void* d_ws, size_t ws_size,
                              hipStream_t stream) {
    (void)in_sizes; (void)n_in; (void)out_size; (void)ws_size;

    const float* x     = (const float*)d_in[0];   // [B, T, 32]
    const float* Wih1f = (const float*)d_in[1];   // [256, 32]
    const float* Whh1f = (const float*)d_in[2];   // [256, 64]
    const float* bih1  = (const float*)d_in[3];
    const float* bhh1  = (const float*)d_in[4];
    const float* Wih2f = (const float*)d_in[5];   // [512, 64]
    const float* Whh2f = (const float*)d_in[6];   // [512, 128]
    const float* bih2  = (const float*)d_in[7];
    const float* bhh2  = (const float*)d_in[8];
    const float* W1    = (const float*)d_in[9];   // [256, 128]
    const float* b1    = (const float*)d_in[10];
    const float* W2    = (const float*)d_in[11];  // [1, 256]
    const float* b2    = (const float*)d_in[12];

    // workspace carve-out (256B aligned); big streams last
    char* wp = (char*)d_ws;
    auto carve = [&wp](size_t bytes) -> void* {
        void* p = (void*)wp;
        wp += (bytes + 255) & ~(size_t)255;
        return p;
    };
    u16*   wih1   = (u16*)  carve((size_t)4 * H1c * DINc * sizeof(u16));
    u16*   whh1   = (u16*)  carve((size_t)4 * H1c * H1c  * sizeof(u16));
    u16*   wih2   = (u16*)  carve((size_t)4 * H2c * H1c  * sizeof(u16));
    u16*   whh2   = (u16*)  carve((size_t)4 * H2c * H2c  * sizeof(u16));
    float* bias1  = (float*)carve((size_t)4 * H1c * sizeof(float));
    float* bias2  = (float*)carve((size_t)4 * H2c * sizeof(float));
    float* h2last = (float*)carve((size_t)Bc * H2c * sizeof(float));
    u16*   xt     = (u16*)  carve((size_t)Tc * Bc * DINc * sizeof(u16)); //  32 MB
    u16*   h1     = (u16*)  carve((size_t)Tc * Bc * H1c  * sizeof(u16)); //  64 MB
    u16*   xg2    = (u16*)  carve((size_t)Tc * Bc * 4 * H2c * sizeof(u16)); // 512 MB

    // prep
    cvt_bf16_kernel<<<32,  256, 0, stream>>>(Wih1f, wih1, 4 * H1c * DINc);
    cvt_bf16_kernel<<<64,  256, 0, stream>>>(Whh1f, whh1, 4 * H1c * H1c);
    cvt_bf16_kernel<<<128, 256, 0, stream>>>(Wih2f, wih2, 4 * H2c * H1c);
    cvt_bf16_kernel<<<256, 256, 0, stream>>>(Whh2f, whh2, 4 * H2c * H2c);
    add_bias_kernel<<<1, 256, 0, stream>>>(bih1, bhh1, bias1, 4 * H1c);
    add_bias_kernel<<<2, 256, 0, stream>>>(bih2, bhh2, bias2, 4 * H2c);
    xpose_cvt_kernel<<<2048, 256, 0, stream>>>(x, xt);

    // layer 1 recurrence: xt -> h1
    lstm1_kernel<<<NB, NTHREADS, 0, stream>>>(xt, wih1, whh1, bias1, h1);

    // input projection for layer 2 (parallel over all B*T rows)
    xg2_gemm_kernel<<<Tc * Bc / MT, NTHREADS, 0, stream>>>(h1, wih2, bias2, xg2);

    // layer 2 recurrence: xg2 -> h2last
    lstm2_kernel<<<NB, NTHREADS, 0, stream>>>(xg2, whh2, h2last);

    // head: y[:, -1]
    head_kernel<<<Bc, H3c, 0, stream>>>(h2last, W1, b1, W2, b2, (float*)d_out);
}
